// MetapathAggrNet_75419625718380
// MI455X (gfx1250) — matrix operand order repaired and checked by reading, not compiled, not run on hardware
//
#include <hip/hip_runtime.h>
#include <hip/hip_bf16.h>
#include <math.h>

// ---------------- problem constants ----------------
#define MP     3        // metapaths
#define AI     4        // instances per example
#define BB     2048     // examples
#define LL     7        // path length (edges)
#define HH     256      // hidden
#define H2     512      // 2*hidden
#define NR     8192     // AI*BB rows per metapath
#define ROWS1  65536    // NR*(LL+1) rows for embed GEMM
#define QROWS  57344    // NR*LL rows for attn GEMM

typedef _Float16 v16h __attribute__((ext_vector_type(16)));
typedef _Float16 v8h  __attribute__((ext_vector_type(8)));
typedef float    v8f  __attribute__((ext_vector_type(8)));

// ---------------- WMMA helpers (CDNA5 16x16x32 f16, f32 accum) ----------------
__device__ __forceinline__ v8f wmma16(v16h a, v16h b, v8f c) {
  // 8 args: (neg_a, A, neg_b, B, c_mod, C, reuse_a, reuse_b)
  return __builtin_amdgcn_wmma_f32_16x16x32_f16(false, a, false, b, (short)0, c, false, false);
}

// A fragment: 16x32 f16, row = lane%16; lanes 0-15 hold K 0..7 & 16..23,
// lanes 16-31 hold K 8..15 & 24..31 (ISA 7.12.2).
__device__ __forceinline__ v16h load_a_frag(const _Float16* rowbase, int k0, int lane) {
  int hf = lane >> 4;
  const _Float16* p = rowbase + k0 + hf * 8;
  v8h x0 = *(const v8h*)(p);
  v8h x1 = *(const v8h*)(p + 16);
  v16h r;
#pragma unroll
  for (int i = 0; i < 8; ++i) { r[i] = x0[i]; r[i + 8] = x1[i]; }
  return r;
}

// B fragment: 32x16 f16, col = lane%16; lanes 0-15 hold K 0..7 & 16..23,
// lanes 16-31 hold K 8..15 & 24..31. colbase = lane's column in transposed weight [N][K].
__device__ __forceinline__ v16h load_b_frag(const _Float16* colbase, int k0, int lane) {
  int hf = lane >> 4;
  const _Float16* p = colbase + k0 + hf * 8;
  v8h x0 = *(const v8h*)(p);
  v8h x1 = *(const v8h*)(p + 16);
  v16h r;
#pragma unroll
  for (int i = 0; i < 8; ++i) { r[i] = x0[i]; r[i + 8] = x1[i]; }
  return r;
}

// ---------------- conversion kernels ----------------
__global__ void cvt_f16_kernel(const float* __restrict__ src, _Float16* __restrict__ dst, int n) {
  int i = blockIdx.x * blockDim.x + threadIdx.x;
  if (i < n) dst[i] = (_Float16)src[i];
}

// dst[n*K + k] = (f16) src[k*N + n]  (store weights column-major = [N][K])
__global__ void tcvt_f16_kernel(const float* __restrict__ src, _Float16* __restrict__ dst, int K, int N) {
  int i = blockIdx.x * blockDim.x + threadIdx.x;
  if (i >= K * N) return;
  int n = i % N, k = i / N;
  dst[(size_t)n * K + k] = (_Float16)src[i];
}

__global__ void zero3_kernel(float* __restrict__ s3) {
  if (threadIdx.x < MP) s3[threadIdx.x] = 0.0f;
}

// ---------------- K1: gathered embed GEMM, 16x64 tile per wave ----------------
// hidden = f16(node_emb[tok]) @ scale_W + b ; A-fragments register-resident, reused over 4 B columns
__global__ void embed_gemm_kernel(const int* __restrict__ tok, const _Float16* __restrict__ nodeE,
                                  const _Float16* __restrict__ Wt, const float* __restrict__ bias,
                                  float* __restrict__ out /*[ROWS1][HH]*/) {
  int wg = blockIdx.x * (blockDim.x >> 5) + (threadIdx.x >> 5);
  int lane = threadIdx.x & 31;
  int cg = wg & 3;         // 4 col groups x 64 cols
  int rt = wg >> 2;        // 4096 row tiles
  int mr = lane & 15, hf = lane >> 4;
  int row = rt * 16 + mr;
  int token = tok[row];
  const _Float16* arow = nodeE + (size_t)token * HH;

  v16h af[8];
#pragma unroll
  for (int kt = 0; kt < 8; ++kt) af[kt] = load_a_frag(arow, kt * 32, lane);

  const _Float16* bcol[4];
#pragma unroll
  for (int j = 0; j < 4; ++j) bcol[j] = Wt + (size_t)((cg * 4 + j) * 16 + mr) * HH;

  v8f acc[4] = {{}, {}, {}, {}};
#pragma unroll
  for (int kt = 0; kt < 8; ++kt) {
#pragma unroll
    for (int j = 0; j < 4; ++j)
      acc[j] = wmma16(af[kt], load_b_frag(bcol[j], kt * 32, lane), acc[j]);
  }
#pragma unroll
  for (int j = 0; j < 4; ++j) {
    int col = (cg * 4 + j) * 16 + mr;
    float bv = bias[col];
#pragma unroll
    for (int v = 0; v < 8; ++v)
      out[(size_t)(rt * 16 + v + 8 * hf) * HH + col] = acc[j][v] + bv;
  }
}

// ---------------- K2: RotatE suffix scan, register-resident per (path, channel) ----------------
__global__ void rot_scan_kernel(const int* __restrict__ etok /*[NR][LL]*/, const float* __restrict__ edgeE /*[16][128]*/,
                                float* __restrict__ hidden /*[NR][8][HH]*/, _Float16* __restrict__ hid16) {
  int t = blockIdx.x * blockDim.x + threadIdx.x;   // NR*128 threads
  int n = t >> 7;
  int c = t & 127;
  float* hb = hidden + (size_t)n * 8 * HH;
  float hr[8], hi[8];
#pragma unroll
  for (int j = 0; j < 8; ++j) { hr[j] = hb[j * HH + c]; hi[j] = hb[j * HH + 128 + c]; }
#pragma unroll
  for (int i = LL - 1; i >= 0; --i) {
    int e = etok[n * LL + i];
    float ang = edgeE[e * 128 + c];
    float cr = cosf(ang), ci = sinf(ang);
    float ir = hr[i], ii = hi[i];       // position i unmodified so far == inputs[:, i]
#pragma unroll
    for (int j = 0; j < 8; ++j) {
      if (j > i) {
        float nr = ir + cr * hr[j] - ci * hi[j];
        float ni = ii + ci * hr[j] + cr * hi[j];
        hr[j] = nr; hi[j] = ni;
      }
    }
  }
  _Float16* h16 = hid16 + (size_t)n * 8 * HH;
#pragma unroll
  for (int j = 0; j < 8; ++j) {
    float inv = 1.0f / (float)(1 + j);
    hr[j] *= inv; hi[j] *= inv;
    hb[j * HH + c] = hr[j];        hb[j * HH + 128 + c] = hi[j];
    h16[j * HH + c] = (_Float16)hr[j]; h16[j * HH + 128 + c] = (_Float16)hi[j];
  }
}

// ---------------- K3: fused attn GEMM + tanh + context dot -> scores[q] ----------------
// h_all row q=(n,l): K 0..255 from hidden[n][0], K 256..511 from hidden[n][1+l]
// All 16 A-fragments held in registers across the 32 column tiles (A traffic /32).
__global__ void attn_score_kernel(const _Float16* __restrict__ hidF16, const _Float16* __restrict__ Wt /*[512][512]T*/,
                                  const float* __restrict__ bias, const float* __restrict__ ctx,
                                  float* __restrict__ scores /*[QROWS]*/) {
  __shared__ float red[4][32][8];
  int w = threadIdx.x >> 5;
  int lane = threadIdx.x & 31;
  int rt = blockIdx.x * (blockDim.x >> 5) + w;   // 3584 row tiles
  int mr = lane & 15, hf = lane >> 4;
  int q = rt * 16 + mr;
  int n = q / LL, l = q % LL;
  const _Float16* base0 = hidF16 + (size_t)(n * 8) * HH;
  const _Float16* base1 = hidF16 + (size_t)(n * 8 + 1 + l) * HH;

  v16h afr[16];
#pragma unroll
  for (int kt = 0; kt < 16; ++kt)
    afr[kt] = load_a_frag((kt < 8) ? base0 : base1, (kt & 7) * 32, lane);

  float rowacc[8];
#pragma unroll
  for (int v = 0; v < 8; ++v) rowacc[v] = 0.0f;
  for (int ctile = 0; ctile < H2 / 16; ++ctile) {
    const _Float16* bcol = Wt + (size_t)(ctile * 16 + mr) * H2;
    v8f acc = {};
#pragma unroll
    for (int kt = 0; kt < H2 / 32; ++kt)
      acc = wmma16(afr[kt], load_b_frag(bcol, kt * 32, lane), acc);
    float bv = bias[ctile * 16 + mr], cv = ctx[ctile * 16 + mr];
#pragma unroll
    for (int v = 0; v < 8; ++v) rowacc[v] += tanhf(acc[v] + bv) * cv;
  }
#pragma unroll
  for (int v = 0; v < 8; ++v) red[w][lane][v] = rowacc[v];
  __syncthreads();
  if (lane < 16) {
    int R = lane, rh = R >> 3, rv = R & 7;
    float s = 0.0f;
#pragma unroll
    for (int j = 0; j < 16; ++j) s += red[w][rh * 16 + j][rv];
    scores[rt * 16 + R] = s;
  }
}

// ---------------- K4a: softmax over L=7 ----------------
__global__ void softmax7_kernel(const float* __restrict__ s, float* __restrict__ w) {
  int n = blockIdx.x * blockDim.x + threadIdx.x;
  if (n >= NR) return;
  float v[LL], mx = -1e30f;
#pragma unroll
  for (int l = 0; l < LL; ++l) { v[l] = s[n * LL + l]; mx = fmaxf(mx, v[l]); }
  float sum = 0.0f;
#pragma unroll
  for (int l = 0; l < LL; ++l) { v[l] = expf(v[l] - mx); sum += v[l]; }
  float inv = 1.0f / sum;
#pragma unroll
  for (int l = 0; l < LL; ++l) w[n * LL + l] = v[l] * inv;
}

// ---------------- K4b: weighted pool + relu + sum over A -> hid[b][m][512] ----------------
__global__ void pool_kernel(const float* __restrict__ hidden, const float* __restrict__ wsoft,
                            float* __restrict__ hid32, _Float16* __restrict__ hid16, int m) {
  int b = blockIdx.x;         // 2048
  int c = threadIdx.x;        // 256
  float a1 = 0.0f, a2 = 0.0f;
#pragma unroll
  for (int a = 0; a < AI; ++a) {
    int n = a * BB + b;
    const float* hb = hidden + (size_t)n * 8 * HH;
    float h0 = hb[c];                         // first half: relu(h0) since softmax sums to 1
    float s = 0.0f;
#pragma unroll
    for (int l = 0; l < LL; ++l) s += wsoft[n * LL + l] * hb[(1 + l) * HH + c];
    a1 += fmaxf(h0, 0.0f);
    a2 += fmaxf(s, 0.0f);
  }
  size_t base = ((size_t)b * MP + m) * H2;
  hid32[base + c] = a1;               hid32[base + HH + c] = a2;
  hid16[base + c] = (_Float16)a1;     hid16[base + HH + c] = (_Float16)a2;
}

// ---------------- K5: inter attention GEMM + tanh + ctx dot, mean over b -> scores3[3] ----------------
__global__ void inter_score_kernel(const _Float16* __restrict__ hid16, const _Float16* __restrict__ Wt,
                                   const float* __restrict__ bias, const float* __restrict__ ctx,
                                   float* __restrict__ s3) {
  __shared__ float red[4][32][8];
  int w = threadIdx.x >> 5;
  int lane = threadIdx.x & 31;
  int rt = blockIdx.x * (blockDim.x >> 5) + w;   // 384 row tiles (6144 rows)
  int mr = lane & 15, hf = lane >> 4;
  int q = rt * 16 + mr;
  const _Float16* arow = hid16 + (size_t)q * H2;

  v16h afr[16];
#pragma unroll
  for (int kt = 0; kt < 16; ++kt) afr[kt] = load_a_frag(arow, kt * 32, lane);

  float rowacc[8];
#pragma unroll
  for (int v = 0; v < 8; ++v) rowacc[v] = 0.0f;
  for (int ctile = 0; ctile < H2 / 16; ++ctile) {
    const _Float16* bcol = Wt + (size_t)(ctile * 16 + mr) * H2;
    v8f acc = {};
#pragma unroll
    for (int kt = 0; kt < H2 / 32; ++kt)
      acc = wmma16(afr[kt], load_b_frag(bcol, kt * 32, lane), acc);
    float bv = bias[ctile * 16 + mr], cv = ctx[ctile * 16 + mr];
#pragma unroll
    for (int v = 0; v < 8; ++v) rowacc[v] += tanhf(acc[v] + bv) * cv;
  }
#pragma unroll
  for (int v = 0; v < 8; ++v) red[w][lane][v] = rowacc[v];
  __syncthreads();
  if (lane < 16) {
    int R = lane, rh = R >> 3, rv = R & 7;
    float s = 0.0f;
#pragma unroll
    for (int j = 0; j < 16; ++j) s += red[w][rh * 16 + j][rv];
    int qq = rt * 16 + R;
    atomicAdd(&s3[qq % MP], s * (1.0f / (float)BB));   // mean over batch
  }
}

// ---------------- K6a: out2h[b][c] = sum_mp hid[b][mp][c] * scores3[mp] (as f16) ----------------
__global__ void mix_kernel(const float* __restrict__ hid32, const float* __restrict__ s3,
                           _Float16* __restrict__ out2h) {
  int t = blockIdx.x * blockDim.x + threadIdx.x;   // BB*H2
  int b = t >> 9, c = t & (H2 - 1);
  float v = 0.0f;
#pragma unroll
  for (int mp = 0; mp < MP; ++mp) v += hid32[((size_t)b * MP + mp) * H2 + c] * s3[mp];
  out2h[(size_t)b * H2 + c] = (_Float16)v;
}

// ---------------- K6b: final GEMM out = out2h @ out_W + out_b, 16x64 tile per wave ----------------
__global__ void out_gemm_kernel(const _Float16* __restrict__ A /*[BB][H2]*/, const _Float16* __restrict__ Wt /*[HH][H2]T*/,
                                const float* __restrict__ bias, float* __restrict__ out /*[BB][HH]*/) {
  int wg = blockIdx.x * (blockDim.x >> 5) + (threadIdx.x >> 5);
  int lane = threadIdx.x & 31;
  int cg = wg & 3, rt = wg >> 2;   // 4 col groups x 128 row tiles
  int mr = lane & 15, hf = lane >> 4;
  const _Float16* arow = A + (size_t)(rt * 16 + mr) * H2;

  v16h afr[16];
#pragma unroll
  for (int kt = 0; kt < 16; ++kt) afr[kt] = load_a_frag(arow, kt * 32, lane);

  const _Float16* bcol[4];
#pragma unroll
  for (int j = 0; j < 4; ++j) bcol[j] = Wt + (size_t)((cg * 4 + j) * 16 + mr) * H2;

  v8f acc[4] = {{}, {}, {}, {}};
#pragma unroll
  for (int kt = 0; kt < 16; ++kt) {
#pragma unroll
    for (int j = 0; j < 4; ++j)
      acc[j] = wmma16(afr[kt], load_b_frag(bcol[j], kt * 32, lane), acc[j]);
  }
#pragma unroll
  for (int j = 0; j < 4; ++j) {
    int col = (cg * 4 + j) * 16 + mr;
    float bv = bias[col];
#pragma unroll
    for (int v = 0; v < 8; ++v)
      out[(size_t)(rt * 16 + v + 8 * hf) * HH + col] = acc[j][v] + bv;
  }
}

// ---------------- host launch ----------------
extern "C" void kernel_launch(void* const* d_in, const int* in_sizes, int n_in,
                              void* d_out, int out_size, void* d_ws, size_t ws_size,
                              hipStream_t stream) {
  (void)in_sizes; (void)n_in; (void)out_size; (void)ws_size;
  const int*   tokens    = (const int*)d_in[0];    // [3][4][2048][8]
  const int*   etok      = (const int*)d_in[1];    // [3][4][2048][7]
  const float* node_emb  = (const float*)d_in[2];  // [50000][256]
  const float* edge_emb  = (const float*)d_in[3];  // [16][128]
  const float* scale_W   = (const float*)d_in[4];  // [256][256]
  const float* scale_b   = (const float*)d_in[5];
  const float* attn_W    = (const float*)d_in[6];  // [512][512]
  const float* attn_b    = (const float*)d_in[7];
  const float* intra_ctx = (const float*)d_in[8];
  const float* inter_W   = (const float*)d_in[9];  // [512][512]
  const float* inter_b   = (const float*)d_in[10];
  const float* inter_ctx = (const float*)d_in[11];
  const float* out_W     = (const float*)d_in[12]; // [512][256]
  const float* out_b     = (const float*)d_in[13];
  float* out = (float*)d_out;                      // [2048][256] f32

  // workspace carve (256B aligned)
  char* w = (char*)d_ws;
  size_t off = 0;
  auto take = [&](size_t bytes) { size_t r = off; off += (bytes + 255) & ~(size_t)255; return r; };
  _Float16* node_f16 = (_Float16*)(w + take((size_t)50000 * HH * 2));     // 25.6 MB
  _Float16* scaleWt  = (_Float16*)(w + take((size_t)HH * HH * 2));        // [N=256][K=256]
  _Float16* attnWt   = (_Float16*)(w + take((size_t)H2 * H2 * 2));        // [N=512][K=512]
  _Float16* interWt  = (_Float16*)(w + take((size_t)H2 * H2 * 2));
  _Float16* outWt    = (_Float16*)(w + take((size_t)HH * H2 * 2));        // [N=256][K=512]
  float*    hidden32 = (float*)(w + take((size_t)ROWS1 * HH * 4));        // 67 MB (reused per mp)
  _Float16* hidden16 = (_Float16*)(w + take((size_t)ROWS1 * HH * 2));     // 33.5 MB
  float*    scoresI  = (float*)(w + take((size_t)QROWS * 4));
  float*    wsoft    = (float*)(w + take((size_t)QROWS * 4));
  float*    hid32    = (float*)(w + take((size_t)BB * MP * H2 * 4));      // 12.6 MB
  _Float16* hid16    = (_Float16*)(w + take((size_t)BB * MP * H2 * 2));
  float*    s3       = (float*)(w + take(256));
  _Float16* out2h    = (_Float16*)(w + take((size_t)BB * H2 * 2));

  // weight / table conversions (f16, weights transposed to [N][K])
  cvt_f16_kernel<<<(50000 * HH + 255) / 256, 256, 0, stream>>>(node_emb, node_f16, 50000 * HH);
  tcvt_f16_kernel<<<(HH * HH + 255) / 256, 256, 0, stream>>>(scale_W, scaleWt, HH, HH);
  tcvt_f16_kernel<<<(H2 * H2 + 255) / 256, 256, 0, stream>>>(attn_W, attnWt, H2, H2);
  tcvt_f16_kernel<<<(H2 * H2 + 255) / 256, 256, 0, stream>>>(inter_W, interWt, H2, H2);
  tcvt_f16_kernel<<<(H2 * HH + 255) / 256, 256, 0, stream>>>(out_W, outWt, H2, HH);
  zero3_kernel<<<1, 32, 0, stream>>>(s3);

  for (int m = 0; m < MP; ++m) {
    // 4096 row tiles x 4 col groups = 16384 waves, 4 waves/block
    embed_gemm_kernel<<<4096, 128, 0, stream>>>(
        tokens + (size_t)m * ROWS1, node_f16, scaleWt, scale_b, hidden32);
    rot_scan_kernel<<<NR * 128 / 256, 256, 0, stream>>>(
        etok + (size_t)m * NR * LL, edge_emb, hidden32, hidden16);
    attn_score_kernel<<<(QROWS / 16) / 4, 128, 0, stream>>>(
        hidden16, attnWt, attn_b, intra_ctx, scoresI);
    softmax7_kernel<<<(NR + 255) / 256, 256, 0, stream>>>(scoresI, wsoft);
    pool_kernel<<<BB, HH, 0, stream>>>(hidden32, wsoft, hid32, hid16, m);
  }

  inter_score_kernel<<<(BB * MP / 16) / 4, 128, 0, stream>>>(hid16, interWt, inter_b, inter_ctx, s3);
  mix_kernel<<<(BB * H2) / 512, 512, 0, stream>>>(hid32, s3, out2h);
  // 128 row tiles x 4 col groups = 512 waves, 4 waves/block
  out_gemm_kernel<<<128, 128, 0, stream>>>(out2h, outWt, out_b, out);
}
